// UnaryLinear_16947940950703
// MI455X (gfx1250) — 8-bit matrix operand order repaired and checked by reading, compiled, not run on hardware
//
#include <hip/hip_runtime.h>
#include <hip/hip_bf16.h>

#define BATCH 8192
#define INF   4096
#define OUTF  4096
#define LRNG  256
#define ACC_BOUND 4097.0f

typedef __attribute__((ext_vector_type(8))) int v8i;
typedef __attribute__((__vector_size__(16))) int vi4;

#define AS1 __attribute__((address_space(1)))
#define AS3 __attribute__((address_space(3)))

#ifndef __has_builtin
#define __has_builtin(x) 0
#endif

#if __has_builtin(__builtin_amdgcn_global_load_async_to_lds_b128) && \
    __has_builtin(__builtin_amdgcn_s_wait_asynccnt)
#define USE_ASYNC_LDS 1
#else
#define USE_ASYNC_LDS 0
#endif

__global__ __launch_bounds__(256) void prep_x_kernel(const float* __restrict__ x,
                                                     signed char* __restrict__ Xd,
                                                     int n4) {
    int i = blockIdx.x * blockDim.x + threadIdx.x;
    if (i < n4) {
        float4 v = ((const float4*)x)[i];
        char4 r;
        r.x = (v.x > 0.5f) ? 1 : 0;
        r.y = (v.y > 0.5f) ? 1 : 0;
        r.z = (v.z > 0.5f) ? 1 : 0;
        r.w = (v.w > 0.5f) ? 1 : 0;
        ((char4*)Xd)[i] = r;
    }
}

__global__ __launch_bounds__(256) void prep_w_kernel(const float* __restrict__ buf_wght,
                                                     const float* __restrict__ buf_bias,
                                                     const float* __restrict__ rng,
                                                     const int*   __restrict__ idx,
                                                     const int*   __restrict__ bias_idx,
                                                     const int*   __restrict__ idx_inv,
                                                     signed char* __restrict__ Wd,
                                                     int*         __restrict__ c) {
    __shared__ float srng[LRNG];
    __shared__ int   red[256];
    const int o = blockIdx.x;
    const int t = threadIdx.x;
    if (t < LRNG) srng[t] = rng[t];
    __syncthreads();

    const float* wrow  = buf_wght + (size_t)o * INF;
    const int*   irow  = idx      + (size_t)o * INF;
    const int*   iirow = idx_inv  + (size_t)o * INF;
    signed char* wd    = Wd       + (size_t)o * INF;

    int cnt = 0;
    for (int i = t; i < INF; i += 256) {
        float wv   = wrow[i];
        int   wbit = (wv > srng[irow[i]  & (LRNG - 1)]) ? 1 : 0;
        int   winv = (wv > srng[iirow[i] & (LRNG - 1)]) ? 0 : 1;
        wd[i] = (signed char)(wbit - winv);
        cnt += winv;
    }
    red[t] = cnt;
    __syncthreads();
    for (int s = 128; s > 0; s >>= 1) {
        if (t < s) red[t] += red[t + s];
        __syncthreads();
    }
    if (t == 0) {
        int bb = (buf_bias[o] > srng[bias_idx[o] & (LRNG - 1)]) ? 1 : 0;
        c[o] = red[0] + bb;
    }
}

#define BM 128
#define BN 128
#define BK 64
#define LDSTRIDE 80
#define ROWQ (LDSTRIDE / 16)
#define NKSTEPS (INF / BK)

__global__ __launch_bounds__(256) void unary_gemm_kernel(const signed char* __restrict__ Xd,
                                                         const signed char* __restrict__ Wd,
                                                         const int*   __restrict__ c,
                                                         const float* __restrict__ accum,
                                                         float*       __restrict__ out) {
    __shared__ int4 sAq[2][BM * ROWQ];
    __shared__ int4 sBq[2][BN * ROWQ];

    const int bn   = blockIdx.x * BN;
    const int bm   = blockIdx.y * BM;
    const int tid  = threadIdx.x;
    const int wave = tid >> 5;
    const int lane = tid & 31;
    const int wm   = wave >> 2;
    const int wn   = wave & 3;
    const int l15  = lane & 15;
    const int kh   = lane >> 4;

    const int r0 = tid >> 2,         q0 = tid & 3;
    const int r1 = (tid + 256) >> 2, q1 = tid & 3;
    const signed char* pA0 = Xd + (size_t)(bm + r0) * INF + q0 * 16;
    const signed char* pA1 = Xd + (size_t)(bm + r1) * INF + q1 * 16;
    const signed char* pB0 = Wd + (size_t)(bn + r0) * INF + q0 * 16;
    const signed char* pB1 = Wd + (size_t)(bn + r1) * INF + q1 * 16;
    const int la0 = r0 * ROWQ + q0;
    const int la1 = r1 * ROWQ + q1;

    v8i acc[4][2];
    v8i afr[4];
    v8i bfr[2];
    const v8i vzero = {0, 0, 0, 0, 0, 0, 0, 0};
#pragma unroll
    for (int mf = 0; mf < 4; ++mf)
#pragma unroll
        for (int nf = 0; nf < 2; ++nf) acc[mf][nf] = vzero;

    #define LOAD_FRAGS(buf)                                                       \
        do {                                                                      \
            const long long* sA8  = (const long long*)&sAq[(buf)][0];             \
            const int4*      sB16 = (const int4*)&sBq[(buf)][0];                  \
            _Pragma("unroll")                                                     \
            for (int nf = 0; nf < 2; ++nf) {                                      \
                int wrow = wn * 32 + nf * 16 + l15;                               \
                union { v8i v; int4 q[2]; } bu;                                   \
                bu.q[0] = sB16[(wrow * LDSTRIDE + kh * 16) >> 4];                 \
                bu.q[1] = sB16[(wrow * LDSTRIDE + 32 + kh * 16) >> 4];            \
                bfr[nf] = bu.v;                                                   \
            }                                                                     \
            _Pragma("unroll")                                                     \
            for (int mf = 0; mf < 4; ++mf) {                                      \
                int arow = wm * 64 + mf * 16 + l15;                               \
                union { v8i v; long long d[4]; } au;                              \
                _Pragma("unroll")                                                 \
                for (int j = 0; j < 4; ++j)                                       \
                    au.d[j] = sA8[(arow * LDSTRIDE + kh * 16 + (j & 1) * 8 + (j >> 1) * 32) >> 3]; \
                afr[mf] = au.v;                                                   \
            }                                                                     \
        } while (0)

    #define DO_WMMAS()                                                            \
        do {                                                                      \
            _Pragma("unroll")                                                     \
            for (int mf = 0; mf < 4; ++mf) {                                      \
                acc[mf][0] = __builtin_amdgcn_wmma_i32_16x16x64_iu8(              \
                    false, afr[mf], true, bfr[0], acc[mf][0], false, false);      \
                acc[mf][1] = __builtin_amdgcn_wmma_i32_16x16x64_iu8(              \
                    false, afr[mf], true, bfr[1], acc[mf][1], false, false);      \
            }                                                                     \
        } while (0)

#if USE_ASYNC_LDS
    #define ISSUE_TILE(buf)                                                    \
        do {                                                                   \
            __builtin_amdgcn_global_load_async_to_lds_b128(                    \
                (AS1 vi4*)pA0, (AS3 vi4*)&sAq[(buf)][la0], 0, 0);              \
            __builtin_amdgcn_global_load_async_to_lds_b128(                    \
                (AS1 vi4*)pA1, (AS3 vi4*)&sAq[(buf)][la1], 0, 0);              \
            __builtin_amdgcn_global_load_async_to_lds_b128(                    \
                (AS1 vi4*)pB0, (AS3 vi4*)&sBq[(buf)][la0], 0, 0);              \
            __builtin_amdgcn_global_load_async_to_lds_b128(                    \
                (AS1 vi4*)pB1, (AS3 vi4*)&sBq[(buf)][la1], 0, 0);              \
            pA0 += BK;  pA1 += BK;  pB0 += BK;  pB1 += BK;                     \
        } while (0)

    ISSUE_TILE(0);
    for (int k = 0; k < NKSTEPS; k += 2) {
        __builtin_amdgcn_s_wait_asynccnt(0);
        __syncthreads();
        LOAD_FRAGS(0);
        ISSUE_TILE(1);
        DO_WMMAS();
        __builtin_amdgcn_s_wait_asynccnt(0);
        __syncthreads();
        LOAD_FRAGS(1);
        ISSUE_TILE(0);
        DO_WMMAS();
    }
#else
    int4 ra0, ra1, rb0, rb1;
    ra0 = *(const int4*)pA0;  ra1 = *(const int4*)pA1;
    rb0 = *(const int4*)pB0;  rb1 = *(const int4*)pB1;
    sAq[0][la0] = ra0;  sAq[0][la1] = ra1;
    sBq[0][la0] = rb0;  sBq[0][la1] = rb1;
    pA0 += BK;  pA1 += BK;  pB0 += BK;  pB1 += BK;
    for (int k = 0; k < NKSTEPS; k += 2) {
        __syncthreads();
        LOAD_FRAGS(0);
        ra0 = *(const int4*)pA0;  ra1 = *(const int4*)pA1;
        rb0 = *(const int4*)pB0;  rb1 = *(const int4*)pB1;
        pA0 += BK;  pA1 += BK;  pB0 += BK;  pB1 += BK;
        DO_WMMAS();
        sAq[1][la0] = ra0;  sAq[1][la1] = ra1;
        sBq[1][la0] = rb0;  sBq[1][la1] = rb1;
        __syncthreads();
        LOAD_FRAGS(1);
        ra0 = *(const int4*)pA0;  ra1 = *(const int4*)pA1;
        rb0 = *(const int4*)pB0;  rb1 = *(const int4*)pB1;
        pA0 += BK;  pA1 += BK;  pB0 += BK;  pB1 += BK;
        DO_WMMAS();
        sAq[0][la0] = ra0;  sAq[0][la1] = ra1;
        sBq[0][la0] = rb0;  sBq[0][la1] = rb1;
    }
#endif

    const int n0 = lane & 15;
    const int mb = (lane >> 4) * 8;
#pragma unroll
    for (int mf = 0; mf < 4; ++mf) {
#pragma unroll
        for (int nf = 0; nf < 2; ++nf) {
            int gn = bn + wn * 32 + nf * 16 + n0;
            int cv = c[gn];
#pragma unroll
            for (int r = 0; r < 8; ++r) {
                int gm = bm + wm * 64 + mf * 16 + mb + r;
                size_t off = (size_t)gm * OUTF + gn;
                float v = accum[off] + (float)(acc[mf][nf][r] + cv);
                out[off] = (v >= ACC_BOUND) ? 1.0f : 0.0f;
            }
        }
    }
}

extern "C" void kernel_launch(void* const* d_in, const int* in_sizes, int n_in,
                              void* d_out, int out_size, void* d_ws, size_t ws_size,
                              hipStream_t stream) {
    const float* x                = (const float*)d_in[0];
    const float* buf_wght         = (const float*)d_in[1];
    const float* buf_bias         = (const float*)d_in[2];
    const float* rng              = (const float*)d_in[3];
    const int*   rng_wght_idx     = (const int*)d_in[4];
    const int*   rng_bias_idx     = (const int*)d_in[5];
    const int*   rng_wght_idx_inv = (const int*)d_in[6];
    const float* accumulator      = (const float*)d_in[7];
    float*       out              = (float*)d_out;

    char* ws = (char*)d_ws;
    signed char* Xd = (signed char*)ws;
    signed char* Wd = (signed char*)(ws + (size_t)BATCH * INF);
    int*         c  = (int*)(ws + (size_t)BATCH * INF + (size_t)OUTF * INF);

    int n4 = BATCH * INF / 4;
    prep_x_kernel<<<(n4 + 255) / 256, 256, 0, stream>>>(x, Xd, n4);
    prep_w_kernel<<<OUTF, 256, 0, stream>>>(buf_wght, buf_bias, rng,
                                            rng_wght_idx, rng_bias_idx,
                                            rng_wght_idx_inv, Wd, c);
    unary_gemm_kernel<<<dim3(OUTF / BN, BATCH / BM), 256, 0, stream>>>(
        Xd, Wd, c, accumulator, out);
}
